// Decoder_50491635532088
// MI455X (gfx1250) — matrix operand order repaired and checked by reading, not compiled, not run on hardware
//
#include <hip/hip_runtime.h>
#include <hip/hip_bf16.h>

typedef __bf16 bf16_t;
typedef __bf16 v16bf __attribute__((ext_vector_type(16)));
typedef float  v8f   __attribute__((ext_vector_type(8)));

union V16BF { v16bf v; bf16_t e[16]; };
union V8F   { v8f  v; float  e[8];  };

#define Bsz       64
#define Tlen      512
#define Hd        512
#define Vocab     50257
#define VocabPad  50304
#define STEPS     27
#define XW        1536   // x_full = [e_t | ctx | h]

__device__ __forceinline__ float sigf(float x) { return 1.0f / (1.0f + __expf(-x)); }

// ---------------------------------------------------------------------------
// f32 -> bf16 conversion with optional zero padding tail (for W_out pad rows)
// ---------------------------------------------------------------------------
__global__ void cvt_pad_kernel(const float* __restrict__ src, bf16_t* __restrict__ dst,
                               long nsrc, long ntot) {
  long stride = (long)gridDim.x * blockDim.x;
  for (long i = (long)blockIdx.x * blockDim.x + threadIdx.x; i < ntot; i += stride)
    dst[i] = (i < nsrc) ? (bf16_t)src[i] : (bf16_t)0.0f;
}

// ---------------------------------------------------------------------------
// state init: h (bf16, into x_buf h-slot), c (f32 copy), y_cur = bos
// ---------------------------------------------------------------------------
__global__ void init_kernel(const float* __restrict__ h0, const float* __restrict__ c0,
                            const int* __restrict__ bos, bf16_t* __restrict__ x_buf,
                            float* __restrict__ c, int* __restrict__ y_cur) {
  int i = blockIdx.x * 256 + threadIdx.x;          // 64*512
  int b = i >> 9, h = i & 511;
  x_buf[b * XW + 1024 + h] = (bf16_t)h0[i];
  c[i] = c0[i];
  if (i < Bsz) y_cur[i] = bos[0];
}

// ---------------------------------------------------------------------------
// Generic WMMA GEMM:  D[m][n] = sum_k A[m][k] * W[n][k]  (+bias0[n]+bias1[n])
//   W[n][k] = (k < ksplit) ? W0[n][k] : W1[n][k-ksplit]
//   block = 8 waves; wave -> 16 N cols, 4 M-tiles (64 rows / blockIdx.y supertile)
// ---------------------------------------------------------------------------
template <bool A_F32>
__global__ void wmma_nt_kernel(const void* __restrict__ Aptr, int lda,
                               const bf16_t* __restrict__ W0, int ldw0, int ksplit,
                               const bf16_t* __restrict__ W1, int ldw1,
                               const float* __restrict__ bias0,
                               const float* __restrict__ bias1,
                               float* __restrict__ D, int ldd, int K) {
  const int lane  = threadIdx.x & 31;
  const int wave  = threadIdx.x >> 5;
  const int nbase = blockIdx.x * 128 + wave * 16;
  const int mrow0 = blockIdx.y * 64;
  const int half  = lane >> 4;
  const int l16   = lane & 15;

  V8F acc[4];
#pragma unroll
  for (int mt = 0; mt < 4; ++mt)
#pragma unroll
    for (int r = 0; r < 8; ++r) acc[mt].e[r] = 0.0f;

  for (int k0 = 0; k0 < K; k0 += 32) {
    // B tile: 32x16, lane = (half,n); 16 contiguous K per lane -> b128 loads
    V16BF bm;
    {
      int kb = k0 + half * 16;
      const bf16_t* wp = (kb < ksplit)
                             ? (W0 + (size_t)(nbase + l16) * ldw0 + kb)
                             : (W1 + (size_t)(nbase + l16) * ldw1 + (kb - ksplit));
#pragma unroll
      for (int e = 0; e < 16; ++e) bm.e[e] = wp[e];
    }
#pragma unroll
    for (int mt = 0; mt < 4; ++mt) {
      V16BF am;
      size_t rb = (size_t)(mrow0 + mt * 16 + l16) * lda;
      int kb0 = k0 + half * 16;       // contiguous K: 16 elems per lane half
      int kb1 = k0 + half * 16 + 8;
      if (A_F32) {
        const float* A = (const float*)Aptr;
#pragma unroll
        for (int e = 0; e < 8; ++e) am.e[e]     = (bf16_t)A[rb + kb0 + e];
#pragma unroll
        for (int e = 0; e < 8; ++e) am.e[8 + e] = (bf16_t)A[rb + kb1 + e];
      } else {
        const bf16_t* A = (const bf16_t*)Aptr;
#pragma unroll
        for (int e = 0; e < 8; ++e) am.e[e]     = A[rb + kb0 + e];
#pragma unroll
        for (int e = 0; e < 8; ++e) am.e[8 + e] = A[rb + kb1 + e];
      }
      acc[mt].v = __builtin_amdgcn_wmma_f32_16x16x32_bf16(
          false, am.v, false, bm.v, (short)0, acc[mt].v, false, false);
    }
  }

  int n = nbase + l16;
  float bv = 0.0f;
  if (bias0) bv += bias0[n];
  if (bias1) bv += bias1[n];
#pragma unroll
  for (int mt = 0; mt < 4; ++mt)
#pragma unroll
    for (int r = 0; r < 8; ++r)
      D[(size_t)(mrow0 + mt * 16 + r + 8 * half) * ldd + n] = acc[mt].e[r] + bv;
}

// ---------------------------------------------------------------------------
// Embedding gather: e_t = emb[y] -> bf16 into x_buf[:, 0:512]
// ---------------------------------------------------------------------------
__global__ void embed_kernel(const float* __restrict__ emb, const int* __restrict__ y_cur,
                             bf16_t* __restrict__ x_buf) {
  int i = blockIdx.x * 256 + threadIdx.x;          // 64*512
  int b = i >> 9, h = i & 511;
  int y = y_cur[b];
  x_buf[b * XW + h] = (bf16_t)emb[(size_t)y * Hd + h];
}

// ---------------------------------------------------------------------------
// Attention scores: one wave per (b,t): v . tanh(Wh_enc[b,t,:] + s[b,:]) (+mask)
// ---------------------------------------------------------------------------
__global__ void scores_kernel(const float* __restrict__ Wh_enc, const float* __restrict__ s,
                              const float* __restrict__ v, const int* __restrict__ mask,
                              float* __restrict__ scores) {
  int wid  = (blockIdx.x * blockDim.x + threadIdx.x) >> 5;   // 0..32767
  int lane = threadIdx.x & 31;
  if (wid >= Bsz * Tlen) return;
  int b = wid >> 9;
  const float* row = Wh_enc + (size_t)wid * Hd;
  const float* sb  = s + b * Hd;
  float sum = 0.0f;
  for (int h = lane; h < Hd; h += 32) sum += v[h] * tanhf(row[h] + sb[h]);
  for (int off = 16; off > 0; off >>= 1) sum += __shfl_down(sum, off, 32);
  if (lane == 0) scores[wid] = mask[wid] ? sum : -1.0e9f;
}

// ---------------------------------------------------------------------------
// Softmax over T + ctx = a . enc_seq ; ctx -> bf16 x_buf[:, 512:1024]
// one block (256 thr) per batch row
// ---------------------------------------------------------------------------
__global__ void softmax_ctx_kernel(const float* __restrict__ scores,
                                   const float* __restrict__ enc_seq,
                                   bf16_t* __restrict__ x_buf) {
  __shared__ float aw[Tlen];
  __shared__ float red[256];
  int b = blockIdx.x, tid = threadIdx.x;
  float v0 = scores[b * Tlen + tid];
  float v1 = scores[b * Tlen + 256 + tid];
  red[tid] = fmaxf(v0, v1);
  __syncthreads();
  for (int s2 = 128; s2 > 0; s2 >>= 1) {
    if (tid < s2) red[tid] = fmaxf(red[tid], red[tid + s2]);
    __syncthreads();
  }
  float M = red[0];
  __syncthreads();
  float e0 = __expf(v0 - M), e1 = __expf(v1 - M);
  red[tid] = e0 + e1;
  __syncthreads();
  for (int s2 = 128; s2 > 0; s2 >>= 1) {
    if (tid < s2) red[tid] += red[tid + s2];
    __syncthreads();
  }
  float inv = 1.0f / red[0];
  aw[tid] = e0 * inv;
  aw[tid + 256] = e1 * inv;
  __syncthreads();
  const float* eb = enc_seq + (size_t)b * Tlen * Hd;
  float a0 = 0.0f, a1 = 0.0f;
  for (int t = 0; t < Tlen; ++t) {
    float at = aw[t];
    a0 += at * eb[(size_t)t * Hd + tid];
    a1 += at * eb[(size_t)t * Hd + 256 + tid];
  }
  x_buf[b * XW + 512 + tid]       = (bf16_t)a0;
  x_buf[b * XW + 512 + 256 + tid] = (bf16_t)a1;
}

// ---------------------------------------------------------------------------
// LSTM cell elementwise; also zeroes argmax accumulator for the logits pass
// gate order (torch): [i | f | g | o] chunks of H
// ---------------------------------------------------------------------------
__global__ void cell_kernel(const float* __restrict__ gates, float* __restrict__ c,
                            bf16_t* __restrict__ x_buf, unsigned long long* __restrict__ gmax) {
  int i = blockIdx.x * 256 + threadIdx.x;          // 64*512
  int b = i >> 9, h = i & 511;
  const float* g = gates + (size_t)b * 2048;
  float ig = sigf(g[h]);
  float fg = sigf(g[512 + h]);
  float gg = tanhf(g[1024 + h]);
  float og = sigf(g[1536 + h]);
  float c2 = fg * c[i] + ig * gg;
  float h2 = og * tanhf(c2);
  c[i] = c2;
  x_buf[b * XW + 1024 + h] = (bf16_t)h2;
  if (i < Bsz) gmax[i] = 0ull;                     // runs before logits kernel
}

// ---------------------------------------------------------------------------
// Logits WMMA fused with argmax: packed u64 (orderable f32 key << 32) | (~n)
// LDS atomicMax pre-reduce, then 64 global atomicMax per block.
// ---------------------------------------------------------------------------
__global__ void logits_argmax_kernel(const bf16_t* __restrict__ h, int lda,
                                     const bf16_t* __restrict__ Wout,
                                     const float* __restrict__ bout,
                                     unsigned long long* __restrict__ gmax) {
  const int lane  = threadIdx.x & 31;
  const int wave  = threadIdx.x >> 5;
  const int nbase = blockIdx.x * 128 + wave * 16;
  const int half  = lane >> 4;
  const int l16   = lane & 15;

  __shared__ unsigned long long smax[Bsz];
  if (threadIdx.x < Bsz) smax[threadIdx.x] = 0ull;
  __syncthreads();

  V8F acc[4];
#pragma unroll
  for (int mt = 0; mt < 4; ++mt)
#pragma unroll
    for (int r = 0; r < 8; ++r) acc[mt].e[r] = 0.0f;

  for (int k0 = 0; k0 < Hd; k0 += 32) {
    V16BF bm;
    const bf16_t* wp = Wout + (size_t)(nbase + l16) * Hd + k0 + half * 16;
#pragma unroll
    for (int e = 0; e < 16; ++e) bm.e[e] = wp[e];
#pragma unroll
    for (int mt = 0; mt < 4; ++mt) {
      V16BF am;
      size_t rb = (size_t)(mt * 16 + l16) * lda;
#pragma unroll
      for (int e = 0; e < 8; ++e) am.e[e]     = h[rb + k0 + half * 16 + e];
#pragma unroll
      for (int e = 0; e < 8; ++e) am.e[8 + e] = h[rb + k0 + half * 16 + 8 + e];
      acc[mt].v = __builtin_amdgcn_wmma_f32_16x16x32_bf16(
          false, am.v, false, bm.v, (short)0, acc[mt].v, false, false);
    }
  }

  int n = nbase + l16;
  if (n < Vocab) {
    float bv = bout[n];
#pragma unroll
    for (int mt = 0; mt < 4; ++mt)
#pragma unroll
      for (int r = 0; r < 8; ++r) {
        float val = acc[mt].e[r] + bv;
        unsigned u = __float_as_uint(val);
        unsigned key = (u & 0x80000000u) ? ~u : (u | 0x80000000u);
        unsigned long long pk =
            ((unsigned long long)key << 32) |
            (unsigned long long)(0xFFFFFFFFu - (unsigned)n);   // ties -> lowest n
        atomicMax(&smax[mt * 16 + r + 8 * half], pk);
      }
  }
  __syncthreads();
  if (threadIdx.x < Bsz) atomicMax(&gmax[threadIdx.x], smax[threadIdx.x]);
}

// ---------------------------------------------------------------------------
// Decode packed argmax -> token id; feed next step and write output [B, 27]
// ---------------------------------------------------------------------------
__global__ void finalize_kernel(const unsigned long long* __restrict__ gmax,
                                int* __restrict__ y_cur, int* __restrict__ out, int step) {
  int b = threadIdx.x;
  if (b < Bsz) {
    int n = (int)(0xFFFFFFFFu - (unsigned)(gmax[b] & 0xFFFFFFFFull));
    y_cur[b] = n;
    out[b * STEPS + step] = n;
  }
}

// ---------------------------------------------------------------------------
extern "C" void kernel_launch(void* const* d_in, const int* in_sizes, int n_in,
                              void* d_out, int out_size, void* d_ws, size_t ws_size,
                              hipStream_t stream) {
  (void)in_sizes; (void)n_in; (void)out_size; (void)ws_size;

  const float* enc_seq = (const float*)d_in[0];
  const int*   enc_mask= (const int*)  d_in[1];
  const float* h0      = (const float*)d_in[2];
  const float* c0      = (const float*)d_in[3];
  const int*   bos     = (const int*)  d_in[4];
  const float* emb     = (const float*)d_in[7];
  const float* W_h     = (const float*)d_in[8];
  const float* W_s     = (const float*)d_in[9];
  const float* vvec    = (const float*)d_in[10];
  const float* W_ih    = (const float*)d_in[11];
  const float* W_hh    = (const float*)d_in[12];
  const float* b_ih    = (const float*)d_in[13];
  const float* b_hh    = (const float*)d_in[14];
  const float* W_out   = (const float*)d_in[15];
  const float* b_out   = (const float*)d_in[16];

  // ---- workspace carve (256B aligned) ----
  char* ws = (char*)d_ws;
  size_t off = 0;
  auto carve = [&](size_t bytes) {
    char* p = ws + off;
    off = (off + bytes + 255) & ~(size_t)255;
    return p;
  };
  float*  Wh_enc   = (float*) carve((size_t)Bsz * Tlen * Hd * 4);      // 67 MB
  bf16_t* Wout_bf  = (bf16_t*)carve((size_t)VocabPad * Hd * 2);        // 51.5 MB
  bf16_t* Wih_bf   = (bf16_t*)carve((size_t)2048 * 1024 * 2);
  bf16_t* Whh_bf   = (bf16_t*)carve((size_t)2048 * 512 * 2);
  bf16_t* Ws_bf    = (bf16_t*)carve((size_t)512 * 512 * 2);
  bf16_t* Wh_bf    = (bf16_t*)carve((size_t)512 * 512 * 2);
  bf16_t* x_buf    = (bf16_t*)carve((size_t)Bsz * XW * 2);             // [e|ctx|h]
  float*  c_buf    = (float*) carve((size_t)Bsz * Hd * 4);
  float*  s_buf    = (float*) carve((size_t)Bsz * Hd * 4);
  float*  sc_buf   = (float*) carve((size_t)Bsz * Tlen * 4);
  float*  gates    = (float*) carve((size_t)Bsz * 2048 * 4);
  unsigned long long* gmax = (unsigned long long*)carve(Bsz * 8);
  int*    y_cur    = (int*)   carve(Bsz * 4);

  // ---- one-time: weight conversion, state init, Wh_enc precompute ----
  cvt_pad_kernel<<<4096, 256, 0, stream>>>(W_out, Wout_bf,
      (long)Vocab * Hd, (long)VocabPad * Hd);
  cvt_pad_kernel<<<2048, 256, 0, stream>>>(W_ih, Wih_bf, 2048L*1024, 2048L*1024);
  cvt_pad_kernel<<<1024, 256, 0, stream>>>(W_hh, Whh_bf, 2048L*512, 2048L*512);
  cvt_pad_kernel<<<512, 256, 0, stream>>>(W_s, Ws_bf, 512L*512, 512L*512);
  cvt_pad_kernel<<<512, 256, 0, stream>>>(W_h, Wh_bf, 512L*512, 512L*512);
  init_kernel<<<128, 256, 0, stream>>>(h0, c0, bos, x_buf, c_buf, y_cur);

  // Wh_enc[bt, k'] = sum_h enc[bt, h] * W_h[k', h]   (M=32768, N=512, K=512)
  wmma_nt_kernel<true><<<dim3(4, 512), 256, 0, stream>>>(
      enc_seq, Hd, Wh_bf, Hd, Hd, Wh_bf, Hd, nullptr, nullptr, Wh_enc, Hd, Hd);

  // ---- 27 greedy decode steps ----
  for (int step = 0; step < STEPS; ++step) {
    embed_kernel<<<128, 256, 0, stream>>>(emb, y_cur, x_buf);

    // s = h . W_s^T    (A = x_buf h-slot, stride XW)
    wmma_nt_kernel<false><<<dim3(4, 1), 256, 0, stream>>>(
        x_buf + 1024, XW, Ws_bf, Hd, Hd, Ws_bf, Hd, nullptr, nullptr,
        s_buf, Hd, Hd);

    scores_kernel<<<4096, 256, 0, stream>>>(Wh_enc, s_buf, vvec, enc_mask, sc_buf);
    softmax_ctx_kernel<<<Bsz, 256, 0, stream>>>(sc_buf, enc_seq, x_buf);

    // gates = x_full . [W_ih | W_hh]^T + b_ih + b_hh   (K=1536 split at 1024)
    wmma_nt_kernel<false><<<dim3(16, 1), 256, 0, stream>>>(
        x_buf, XW, Wih_bf, 1024, 1024, Whh_bf, Hd, b_ih, b_hh,
        gates, 2048, 1536);

    cell_kernel<<<128, 256, 0, stream>>>(gates, c_buf, x_buf, gmax);

    logits_argmax_kernel<<<VocabPad / 128, 256, 0, stream>>>(
        x_buf + 1024, XW, Wout_bf, b_out, gmax);

    finalize_kernel<<<1, 64, 0, stream>>>(gmax, y_cur, (int*)d_out, step);
  }
}
